// PureKernalMetricLogits_12240656793881
// MI455X (gfx1250) — matrix operand order repaired and checked by reading, not compiled, not run on hardware
//
#include <hip/hip_runtime.h>
#include <hip/hip_bf16.h>
#include <math.h>

// ---------------------------------------------------------------------------
// PureKernalMetricLogits for MI455X (gfx1250, wave32, WMMA)
//
//   metric(B,C)  = ||x||^2 + ||w||^2 - 2 x.w
//   kernal       = exp(-metric/1.2)
//   scale        = 3*log(C-1)/max(mean(kernal[i,label[i]]), 0.5)
//   out          = scale * kernal            (4096 x 10000 f32)
//
// Roofline: out stores = 164 MB (~7us @ 23.3 TB/s) dominate HBM; GEMM is
// 42 GFLOP -> needs ~6 PFLOPS to stay store-bound -> bf16 WMMA w/ f32 acc.
// bf16 operands (feat 4MB + weights 10MB) fit in the 192MB L2, so block-level
// re-reads never touch HBM; direct global_load_b128 fragment loads suffice.
//
// Pipeline (all on `stream`):
//  1) prep_bf16_rows(feat)    : f32->bf16 convert + row ||.||^2   (x2)
//  2) prep_bf16_rows(weights) : f32->bf16 convert + row ||.||^2   (w2)
//  3) zero1                   : zero the cor accumulator
//  4) cor_kernel              : per-sample dot at label (f32), atomic sum
//  5) scale_kernel            : scalar scale3 = 3*log(9999)/max(mean,0.5)
//  6) gemm_logits             : bf16 WMMA GEMM + fused exp/scale epilogue
// ---------------------------------------------------------------------------

#define BATCH_    4096
#define CLASSES_  10000
#define KDIM_     512

// exp(-m/1.2) = exp2(-m / (1.2*ln2))
#define INV12_LOG2E 1.2022458674074695f

typedef __attribute__((ext_vector_type(16))) __bf16 v16bf;
typedef __attribute__((ext_vector_type(8)))  __bf16 v8bf;
typedef __attribute__((ext_vector_type(4)))  __bf16 v4bf;
typedef __attribute__((ext_vector_type(8)))  float  v8f;

__device__ __forceinline__ __bf16 f2bf(float f) {
  unsigned u = __float_as_uint(f);
  u += 0x7FFFu + ((u >> 16) & 1u);          // round-to-nearest-even
  unsigned short h = (unsigned short)(u >> 16);
  __bf16 r;
  __builtin_memcpy(&r, &h, 2);
  return r;
}

// --- 1/2: convert a 512-wide f32 row to bf16 and emit the row squared-norm --
__global__ __launch_bounds__(128) void prep_bf16_rows(
    const float* __restrict__ src, __bf16* __restrict__ dst,
    float* __restrict__ row_sq) {
  __shared__ float red[4];
  const int row = blockIdx.x;
  const int t   = threadIdx.x;                       // 128 threads, 4 elems each
  const float4 v = ((const float4*)(src + (size_t)row * KDIM_))[t];
  v4bf b;
  b[0] = f2bf(v.x); b[1] = f2bf(v.y); b[2] = f2bf(v.z); b[3] = f2bf(v.w);
  ((v4bf*)(dst + (size_t)row * KDIM_))[t] = b;
  float s = v.x * v.x + v.y * v.y + v.z * v.z + v.w * v.w;
#pragma unroll
  for (int off = 16; off; off >>= 1) s += __shfl_xor(s, off, 32);  // wave32
  if ((t & 31) == 0) red[t >> 5] = s;
  __syncthreads();
  if (t == 0) row_sq[row] = red[0] + red[1] + red[2] + red[3];
}

// --- 3: zero the scalar accumulator -----------------------------------------
__global__ void zero1(float* p) { *p = 0.0f; }

// --- 4: cor[i] = exp(-metric(i,label[i])/1.2) in full f32; atomic mean-sum --
__global__ __launch_bounds__(256) void cor_kernel(
    const float* __restrict__ feat, const float* __restrict__ weights,
    const int* __restrict__ label, const float* __restrict__ x2,
    const float* __restrict__ w2, float* __restrict__ corsum) {
  const int wave = threadIdx.x >> 5;
  const int lane = threadIdx.x & 31;
  const int i    = blockIdx.x * 8 + wave;            // one wave per sample
  const int lab  = label[i];
  const float* fr = feat    + (size_t)i   * KDIM_;
  const float* wr = weights + (size_t)lab * KDIM_;
  float d = 0.0f;
#pragma unroll 4
  for (int j = lane; j < KDIM_; j += 32) d += fr[j] * wr[j];
#pragma unroll
  for (int off = 16; off; off >>= 1) d += __shfl_xor(d, off, 32);
  if (lane == 0) {
    const float metric = x2[i] + w2[lab] - 2.0f * d;
    atomicAdd(corsum, exp2f(-metric * INV12_LOG2E));
  }
}

// --- 5: scalar scale --------------------------------------------------------
__global__ void scale_kernel(const float* __restrict__ corsum,
                             float* __restrict__ s3) {
  const float avg = fmaxf(*corsum * (1.0f / (float)BATCH_), 0.5f);
  *s3 = 3.0f * logf((float)CLASSES_ - 1.0f) / avg;
}

// --- 6: bf16 WMMA GEMM with fused exp/scale epilogue ------------------------
// Block = 256 threads = 8 waves in a 2(M) x 4(N) grid; each wave owns a
// 64x32 output patch = 4x2 tiles of 16x16, accumulated in f32 over K=512
// via v_wmma_f32_16x16x32_bf16 (16 K-steps).
__global__ __launch_bounds__(256) void gemm_logits(
    const __bf16* __restrict__ fb, const __bf16* __restrict__ wb,
    const float* __restrict__ x2, const float* __restrict__ w2,
    const float* __restrict__ s3p, float* __restrict__ out) {
  const int lane = threadIdx.x & 31;
  const int wave = threadIdx.x >> 5;
  const int wm   = wave >> 2;                        // 0..1
  const int wn   = wave & 3;                         // 0..3
  const int hh   = lane >> 4;                        // half-wave: 0/1
  const int l16  = lane & 15;

  const int m0 = blockIdx.y * 128 + wm * 64;
  int nblk = blockIdx.x * 128;
  if (nblk > CLASSES_ - 128) nblk = CLASSES_ - 128;  // clamp tail (overlap OK)
  const int n0 = nblk + wn * 32;

  v8f acc[4][2] = {};

  // ISA bf16 A layout (16x32): lanes0-15 K=0..7 & 16..23, lanes16-31 K=8..15 & 24..31
  const __bf16* aptr[4];
#pragma unroll
  for (int mt = 0; mt < 4; ++mt)
    aptr[mt] = fb + (size_t)(m0 + mt * 16 + l16) * KDIM_ + hh * 8;
  // ISA bf16 B layout (32x16): lanes0-15 K=0..7 & 16..23, lanes16-31 K=8..15 & 24..31
  const __bf16* bptr[2];
#pragma unroll
  for (int nt = 0; nt < 2; ++nt)
    bptr[nt] = wb + (size_t)(n0 + nt * 16 + l16) * KDIM_ + hh * 8;

  for (int k0 = 0; k0 < KDIM_; k0 += 32) {
    // Unconditional speculative prefetch of the next K-chunk: keeps the loop
    // branch-free; the final-iteration address is still inside the workspace
    // allocation, and gfx1250 prefetch drops bad translations silently.
    __builtin_prefetch(aptr[0] + k0 + 32, 0, 1);
    __builtin_prefetch(bptr[0] + k0 + 32, 0, 1);

    v16bf a[4], b[2];
#pragma unroll
    for (int mt = 0; mt < 4; ++mt) {
      v8bf lo = *(const v8bf*)(aptr[mt] + k0);
      v8bf hi = *(const v8bf*)(aptr[mt] + k0 + 16);
      a[mt] = __builtin_shufflevector(lo, hi, 0, 1, 2, 3, 4, 5, 6, 7, 8, 9, 10,
                                      11, 12, 13, 14, 15);
    }
#pragma unroll
    for (int nt = 0; nt < 2; ++nt) {
      v8bf lo = *(const v8bf*)(bptr[nt] + k0);
      v8bf hi = *(const v8bf*)(bptr[nt] + k0 + 16);
      b[nt] = __builtin_shufflevector(lo, hi, 0, 1, 2, 3, 4, 5, 6, 7, 8, 9, 10,
                                      11, 12, 13, 14, 15);
    }
#pragma unroll
    for (int mt = 0; mt < 4; ++mt)
#pragma unroll
      for (int nt = 0; nt < 2; ++nt)
        acc[mt][nt] = __builtin_amdgcn_wmma_f32_16x16x32_bf16(
            false, a[mt], false, b[nt], (short)0, acc[mt][nt], false, false);
  }

  // fused epilogue: out = s3 * exp2(-(x2+w2-2ab) * INV12_LOG2E)
  const float s3 = *s3p;
  float w2v[2];
#pragma unroll
  for (int nt = 0; nt < 2; ++nt) w2v[nt] = w2[n0 + nt * 16 + l16];

#pragma unroll
  for (int mt = 0; mt < 4; ++mt) {
    const int mb = m0 + mt * 16 + hh * 8;            // C/D layout: vgpr r -> M=mb+r
#pragma unroll
    for (int r = 0; r < 8; ++r) {
      const int m = mb + r;
      const float xv = x2[m];
#pragma unroll
      for (int nt = 0; nt < 2; ++nt) {
        const float metric = xv + w2v[nt] - 2.0f * acc[mt][nt][r];
        out[(size_t)m * CLASSES_ + (n0 + nt * 16 + l16)] =
            s3 * exp2f(-metric * INV12_LOG2E);
      }
    }
  }
}

// ---------------------------------------------------------------------------
extern "C" void kernel_launch(void* const* d_in, const int* in_sizes, int n_in,
                              void* d_out, int out_size, void* d_ws,
                              size_t ws_size, hipStream_t stream) {
  const float* feat    = (const float*)d_in[0];   // (4096, 512) f32
  const int*   label   = (const int*)d_in[1];     // (4096,)     i32
  const float* weights = (const float*)d_in[2];   // (10000,512) f32
  float* out = (float*)d_out;                     // (4096,10000) f32

  // workspace layout (~14.5 MB)
  char* ws = (char*)d_ws;
  constexpr size_t OFF_FEATB = 0;                             // 4096*512*2
  constexpr size_t OFF_WB    = (size_t)BATCH_ * KDIM_ * 2;    // 10000*512*2
  constexpr size_t OFF_X2    = OFF_WB + (size_t)CLASSES_ * KDIM_ * 2;
  constexpr size_t OFF_W2    = OFF_X2 + (size_t)BATCH_ * 4;
  constexpr size_t OFF_SUM   = OFF_W2 + (size_t)CLASSES_ * 4;
  constexpr size_t OFF_S3    = OFF_SUM + 4;
  (void)OFF_FEATB;
  __bf16* featb = (__bf16*)(ws + OFF_FEATB);
  __bf16* wb    = (__bf16*)(ws + OFF_WB);
  float*  x2    = (float*)(ws + OFF_X2);
  float*  w2    = (float*)(ws + OFF_W2);
  float*  csum  = (float*)(ws + OFF_SUM);
  float*  s3    = (float*)(ws + OFF_S3);

  prep_bf16_rows<<<BATCH_, 128, 0, stream>>>(feat, featb, x2);
  prep_bf16_rows<<<CLASSES_, 128, 0, stream>>>(weights, wb, w2);
  zero1<<<1, 1, 0, stream>>>(csum);
  cor_kernel<<<BATCH_ / 8, 256, 0, stream>>>(feat, weights, label, x2, w2, csum);
  scale_kernel<<<1, 1, 0, stream>>>(csum, s3);

  dim3 grid((CLASSES_ + 127) / 128, BATCH_ / 128);   // 79 x 32
  gemm_logits<<<grid, 256, 0, stream>>>(featb, wb, x2, w2, s3, out);
}
